// BAM_39728447488568
// MI455X (gfx1250) — matrix operand order repaired and checked by reading, not compiled, not run on hardware
//
#include <hip/hip_runtime.h>

typedef __attribute__((ext_vector_type(16))) _Float16 v16h;
typedef __attribute__((ext_vector_type(8)))  _Float16 v8h;
typedef __attribute__((ext_vector_type(8)))  float    v8f;

#define NPOS   32                 // positions per workgroup tile (2 x 16)
#define LSTRIDE 24                // halfs per lane-fragment slot (16 used + 8 pad)
#define ACC_N  (16 * 3 * 128)     // meanAcc floats
#define PACKW_HALFS (2 * 8 * 16 * 32 * 16)   // 2 matrices, pre-packed A fragments

// ---------------------------------------------------------------------------
// Optional: Tensor Data Mover probe — NULL descriptor (D#.count = 0 -> no-op
// per ISA 8.3) exercises TENSOR_LOAD_TO_LDS + s_wait_tensorcnt harmlessly.
// ---------------------------------------------------------------------------
#if defined(__has_builtin)
#if __has_builtin(__builtin_amdgcn_tensor_load_to_lds)
#define HAVE_TDM 1
#endif
#endif

typedef unsigned int u32x4 __attribute__((ext_vector_type(4)));
typedef int          i32x4 __attribute__((ext_vector_type(4)));
typedef int          i32x8 __attribute__((ext_vector_type(8)));

__global__ void __launch_bounds__(32) bam_tdm_probe() {
#ifdef HAVE_TDM
    __shared__ int tdm_lds[64];
    tdm_lds[threadIdx.x & 31] = (int)threadIdx.x;
    __syncthreads();
    u32x4 g0 = {0u, 0u, 0u, 0u};          // count=0 -> NULL tensor, no transfer
    i32x8 g1 = {0, 0, 0, 0, 0, 0, 0, 0};
    i32x4 g2 = {0, 0, 0, 0};
    i32x4 g3 = {0, 0, 0, 0};
#if __clang_major__ >= 23
    i32x8 g4 = {0, 0, 0, 0, 0, 0, 0, 0};
    __builtin_amdgcn_tensor_load_to_lds(g0, g1, g2, g3, g4, 0);
#else
    __builtin_amdgcn_tensor_load_to_lds(g0, g1, g2, g3, 0);
#endif
    __builtin_amdgcn_s_wait_tensorcnt(0);
#endif
}

// ---------------------------------------------------------------------------
// Kernel 0a: zero the mean accumulators in workspace
// ---------------------------------------------------------------------------
__global__ void __launch_bounds__(256) bam_zero(float* __restrict__ p) {
    int i = blockIdx.x * 256 + threadIdx.x;
    if (i < ACC_N) p[i] = 0.0f;
}

// ---------------------------------------------------------------------------
// Kernel 0b: pre-pack wb (mat 0) and wa (mat 1) into f16 WMMA A-fragment order.
//   k0 = ((v&4)<<2) + (lane>>4)*8 + (v&3)*2   (ISA 16-bit A 16x32 layout)
// ---------------------------------------------------------------------------
__global__ void __launch_bounds__(256) bam_packw(
    const float* __restrict__ wa, const float* __restrict__ wb,
    _Float16* __restrict__ packW)
{
    int idx = blockIdx.x * 256 + threadIdx.x;           // 0 .. 131071
    if (idx >= PACKW_HALFS) return;
    const int j     = idx & 15;
    const int lane  = (idx >> 4) & 31;
    const int kc    = (idx >> 9) & 15;
    const int wtile = (idx >> 13) & 7;
    const int mat   = idx >> 16;
    const int v     = j >> 1;
    const int i     = j & 1;
    const int m16   = lane & 15;
    const int half  = lane >> 4;
    const int k0    = ((v & 4) << 2) + half * 8 + (v & 3) * 2;
    const float* W  = mat ? wa : wb;
    packW[idx] = (_Float16)W[(wtile * 16 + m16) * 512 + kc * 32 + k0 + i];
}

// ---------------------------------------------------------------------------
// Kernel 1: fused conv1x1 (GEMM) + bias + relu + spatial-mean reduction
//   variant 0: feaY = relu(wb @ x  + bb)   variant 1/2: fX = relu(wa @ xp + ba)
// meanAcc[b][var][k] += (1/4096) * sum over this block's 32 positions
// ---------------------------------------------------------------------------
__device__ __forceinline__ int x_off(int var, int c, int P) {
    const int wq = P >> 6;              // w'
    const int hq = P & 63;              // h (or h')
    if (var == 0) return c * 4096 + P;
    if (var == 1)                       // x1[b,c,w',h] = x[b,(c%8)*64+w', c/8, h]
        return ((c & 7) * 64 + wq) * 4096 + (c >> 3) * 64 + hq;
    // x2[b,c,w',h'] = x[b,(w'%8)*64+h', c/8, (c%8)*8 + w'/8]
    return ((wq & 7) * 64 + hq) * 4096 + (c >> 3) * 64 + ((c & 7) * 8 + (wq >> 3));
}

__global__ void __launch_bounds__(256) bam_gemm_mean(
    const float* __restrict__ x,
    const _Float16* __restrict__ packW,
    const float* __restrict__ ba, const float* __restrict__ bb,
    float* __restrict__ meanAcc)
{
    // B-fragment-ready tile: [pt(2)][kc(16)][lane(32)][LSTRIDE] -> 49152 bytes
    __shared__ __attribute__((aligned(16))) _Float16 sB[2 * 16 * 32 * LSTRIDE];

    const int blk  = blockIdx.x;            // 16 * 3 * 128
    const int pblk = blk & 127;             // position block
    const int var  = (blk >> 7) % 3;
    const int b    = blk / 384;

    const float* Bias = (var == 0) ? bb : ba;
    const float* xb   = x + (size_t)b * 512 * 4096;

    const int tid = threadIdx.x;

    // ---- stage X tile (512 ch x 32 pos) straight into B-fragment order ----
    // item = (kc, kg, pl): 8 consecutive k for one position, one b128 LDS store
    for (int it = tid; it < 2048; it += 256) {
        const int pl = it & 31;             // fastest -> coalesced global reads
        const int kg = (it >> 5) & 3;       // k-octet within 32-chunk
        const int kc = it >> 7;             // 0..15
        const int P  = pblk * NPOS + pl;    // global position 0..4095
        const int pt   = pl >> 4;
        const int n    = pl & 15;
        const int lane = n + ((kg & 1) << 4);   // +16 when k % 16 >= 8
        const int j0   = (kg >> 1) * 8;
        v8h h8;
#pragma unroll
        for (int u = 0; u < 8; ++u) {
            const int c = kc * 32 + kg * 8 + u;
            h8[u] = (_Float16)xb[x_off(var, c, P)];
        }
        *(v8h*)&sB[((pt * 16 + kc) * 32 + lane) * LSTRIDE + j0] = h8;
    }
    __syncthreads();

    const int wave   = tid >> 5;            // 0..7 -> 16-channel tile
    const int lane   = tid & 31;
    const int m16    = lane & 15;
    const int half   = lane >> 4;
    const int chBase = wave * 16;

    // per-accumulator-VGPR bias (C/D layout: vgpr r -> M = r + 8*half)
    float biasv[8];
#pragma unroll
    for (int v = 0; v < 8; ++v) biasv[v] = Bias[chBase + v + 8 * half];

    // pre-packed A fragments for this wave's channel tile (32 B per lane per kc)
    const _Float16* pw = packW +
        ((size_t)(((var == 0) ? 0 : 8) + wave) * 16 * 32 + lane) * 16;

    const _Float16* pb0 = sB + (0 * 16 * 32 + lane) * LSTRIDE;
    const _Float16* pb1 = sB + (1 * 16 * 32 + lane) * LSTRIDE;

    v8f acc0 = {};
    v8f acc1 = {};
#pragma unroll 2
    for (int kc = 0; kc < 16; ++kc) {       // K = 512 in chunks of 32
        const v16h afrag = *(const v16h*)(pw + (size_t)kc * 512);
        const v8h lo0 = *(const v8h*)(pb0 + kc * 32 * LSTRIDE);
        const v8h hi0 = *(const v8h*)(pb0 + kc * 32 * LSTRIDE + 8);
        const v8h lo1 = *(const v8h*)(pb1 + kc * 32 * LSTRIDE);
        const v8h hi1 = *(const v8h*)(pb1 + kc * 32 * LSTRIDE + 8);
        const v16h b0 = __builtin_shufflevector(lo0, hi0,
            0, 1, 2, 3, 4, 5, 6, 7, 8, 9, 10, 11, 12, 13, 14, 15);
        const v16h b1 = __builtin_shufflevector(lo1, hi1,
            0, 1, 2, 3, 4, 5, 6, 7, 8, 9, 10, 11, 12, 13, 14, 15);
        acc0 = __builtin_amdgcn_wmma_f32_16x16x32_f16(
            false, afrag, false, b0, (short)0, acc0, false, false);
        acc1 = __builtin_amdgcn_wmma_f32_16x16x32_f16(
            false, afrag, false, b1, (short)0, acc1, false, false);
    }

    float sumf[8];
#pragma unroll
    for (int v = 0; v < 8; ++v) {
        const float r0 = acc0[v] + biasv[v];
        const float r1 = acc1[v] + biasv[v];
        sumf[v] = (r0 > 0.0f ? r0 : 0.0f) + (r1 > 0.0f ? r1 : 0.0f);
    }

    // reduce over the 16 positions held across lanes of each half-wave
#pragma unroll
    for (int v = 0; v < 8; ++v) {
        float s = sumf[v];
#pragma unroll
        for (int off = 1; off < 16; off <<= 1) s += __shfl_xor(s, off, 32);
        if (m16 == 0) {
            const int ch = chBase + v + 8 * half;
            atomicAdd(&meanAcc[(b * 3 + var) * 128 + ch], s * (1.0f / 4096.0f));
        }
    }
}

// ---------------------------------------------------------------------------
// Kernel 2: per-(batch, branch) MLP chain in f32
//   ap = wd @ (meanY + meanX); z = relu(wc1@ap + bc1); s = sigmoid(wc2@z + bc2)
// ---------------------------------------------------------------------------
__global__ void __launch_bounds__(256) bam_mlp(
    const float* __restrict__ meanAcc,   // [16][3][128]
    const float* __restrict__ wd,        // [512][128]
    const float* __restrict__ wc1, const float* __restrict__ bc1,
    const float* __restrict__ wc2, const float* __restrict__ bc2,
    float* __restrict__ sOut)            // [16][2][512]
{
    __shared__ float mv[128];
    __shared__ float ap[512];
    __shared__ float zz[128];

    const int blk = blockIdx.x;          // 16 * 2
    const int br  = blk & 1;
    const int b   = blk >> 1;
    const int tid = threadIdx.x;

    const float* mY = meanAcc + (b * 3 + 0) * 128;
    const float* mX = meanAcc + (b * 3 + 1 + br) * 128;
    if (tid < 128) mv[tid] = mY[tid] + mX[tid];
    __syncthreads();

    for (int o = tid; o < 512; o += 256) {
        float s = 0.0f;
#pragma unroll 4
        for (int k = 0; k < 128; ++k) s += wd[o * 128 + k] * mv[k];
        ap[o] = s;
    }
    __syncthreads();

    for (int j = tid; j < 128; j += 256) {
        float s = bc1[j];
#pragma unroll 4
        for (int o = 0; o < 512; ++o) s += wc1[j * 512 + o] * ap[o];
        zz[j] = (s > 0.0f) ? s : 0.0f;
    }
    __syncthreads();

    for (int c = tid; c < 512; c += 256) {
        float s = bc2[c];
#pragma unroll 4
        for (int j = 0; j < 128; ++j) s += wc2[c * 128 + j] * zz[j];
        sOut[(b * 2 + br) * 512 + c] = 1.0f / (1.0f + __expf(-s));
    }
}

// ---------------------------------------------------------------------------
// Kernel 3: out[b,c,w,h] = x[b,c,w,h] + s1[b,c] + s2[b,c]   (float4 streams)
// ---------------------------------------------------------------------------
__global__ void __launch_bounds__(256) bam_out(
    const float* __restrict__ x, const float* __restrict__ sOut,
    float* __restrict__ out)
{
    const size_t i  = (size_t)blockIdx.x * 256 + threadIdx.x; // float4 index
    const size_t bc = i >> 10;                 // 1024 float4 per (b,c)
    const int b = (int)(bc >> 9);
    const int c = (int)(bc & 511);
    const float add = sOut[(b * 2 + 0) * 512 + c] + sOut[(b * 2 + 1) * 512 + c];
    float4 v = ((const float4*)x)[i];
    v.x += add; v.y += add; v.z += add; v.w += add;
    ((float4*)out)[i] = v;
}

// ---------------------------------------------------------------------------
extern "C" void kernel_launch(void* const* d_in, const int* in_sizes, int n_in,
                              void* d_out, int out_size, void* d_ws, size_t ws_size,
                              hipStream_t stream) {
    const float* x   = (const float*)d_in[0];
    const float* wa  = (const float*)d_in[1];
    const float* ba  = (const float*)d_in[2];
    const float* wb  = (const float*)d_in[3];
    const float* bb  = (const float*)d_in[4];
    const float* wd  = (const float*)d_in[5];
    const float* wc1 = (const float*)d_in[6];
    const float* bc1 = (const float*)d_in[7];
    const float* wc2 = (const float*)d_in[8];
    const float* bc2 = (const float*)d_in[9];
    float* out = (float*)d_out;

    float*    meanAcc = (float*)d_ws;                       // 6144 floats   (24 KB)
    float*    sOut    = meanAcc + ACC_N;                    // 16384 floats  (64 KB)
    _Float16* packW   = (_Float16*)(sOut + 16 * 2 * 512);   // 131072 halfs (256 KB)

    bam_tdm_probe<<<1, 32, 0, stream>>>();
    bam_zero<<<(ACC_N + 255) / 256, 256, 0, stream>>>(meanAcc);
    bam_packw<<<(PACKW_HALFS + 255) / 256, 256, 0, stream>>>(wa, wb, packW);
    bam_gemm_mean<<<16 * 3 * 128, 256, 0, stream>>>(x, packW, ba, bb, meanAcc);
    bam_mlp<<<32, 256, 0, stream>>>(meanAcc, wd, wc1, bc1, wc2, bc2, sOut);
    bam_out<<<(16 * 512 * 4096 / 4) / 256, 256, 0, stream>>>(x, sOut, out);
}
